// xLSTMModel_45844480918225
// MI455X (gfx1250) — matrix operand order repaired and checked by reading, not compiled, not run on hardware
//
#include <hip/hip_runtime.h>
#include <hip/hip_bf16.h>
#include <stdint.h>

// ---------------------------------------------------------------------------
// Problem constants (from reference):  B=64, S=512, D=128, H=512, OUT=24
// ---------------------------------------------------------------------------
#define BB   64
#define SS   512
#define DD   128
#define HH   512
#define OUTN 24

typedef __attribute__((ext_vector_type(16))) _Float16 v16h;
typedef __attribute__((ext_vector_type(8)))  float    v8f;

// ---------------------------------------------------------------------------
// Fast transcendental helpers: v_exp_f32 + v_rcp_f32 (no IEEE divide chains)
// ---------------------------------------------------------------------------
__device__ __forceinline__ float fast_sigmoid(float x) {
    return __builtin_amdgcn_rcpf(1.0f + __expf(-x));
}
__device__ __forceinline__ float fast_tanh(float x) {
    // 1 - 2/(e^{2x}+1): saturates correctly to +/-1 without NaN at extremes
    float e = __expf(2.0f * x);
    return 1.0f - 2.0f * __builtin_amdgcn_rcpf(e + 1.0f);
}

// ---------------------------------------------------------------------------
// f32 -> f16 conversion (grid-stride)
// ---------------------------------------------------------------------------
__global__ void convert_f32_to_f16(const float* __restrict__ src,
                                   _Float16* __restrict__ dst, int n) {
    int i = blockIdx.x * blockDim.x + threadIdx.x;
    int stride = gridDim.x * blockDim.x;
    for (; i < n; i += stride) dst[i] = (_Float16)src[i];
}

__global__ void zero_f16(_Float16* __restrict__ a, _Float16* __restrict__ b, int n) {
    int i = blockIdx.x * blockDim.x + threadIdx.x;
    int stride = gridDim.x * blockDim.x;
    for (; i < n; i += stride) { a[i] = (_Float16)0.0f; b[i] = (_Float16)0.0f; }
}

// ---------------------------------------------------------------------------
// WMMA fragment container: two 16B-contiguous quads per lane
// ---------------------------------------------------------------------------
union Frag {
    v16h v;
    uint4 q[2];
};

// A-fragment (16x32 f16, ISA 7.12.2): lane m = lane%16, half = lane/16.
// Per lane the 16 halves live as two 16-byte contiguous runs at
// (kb + half*8) and (kb + half*8 + 16) elements.
__device__ __forceinline__ Frag load_a_frag(const _Float16* __restrict__ row,
                                            int kb, int half) {
    Frag a;
    const _Float16* s = row + kb + (half << 3);
    a.q[0] = *(const uint4*)(s);
    a.q[1] = *(const uint4*)(s + 16);
    return a;
}

// B-fragment (32x16 f16): lane n = lane%16; same k order as A:
// two 16-byte runs at (kb + half*8) and (kb + half*8 + 16) elements.
__device__ __forceinline__ Frag load_b_frag(const _Float16* __restrict__ row,
                                            int kb, int half) {
    Frag b;
    const _Float16* s = row + kb + (half << 3);
    b.q[0] = *(const uint4*)(s);
    b.q[1] = *(const uint4*)(s + 16);
    return b;
}

// ---------------------------------------------------------------------------
// Fused LSTM step: 4 gate GEMMs (WMMA f16->f32) + cell update, one kernel.
//   A (concat [inA1 | hIn]) : [64, K]  (K = K1 + 512), f16
//   W_g                     : [512, K] row-major, f16 (B matrix = W^T)
//   output: hOut [64,512] f16 (double buffered), cState [64,512] f32 in-place
// Grid: 16 blocks x 256 threads = 128 waves; wave -> one 16x16 tile, 4 accums.
// K-loop is split at K1 (multiple of 32) so each segment has uniform A base.
// ---------------------------------------------------------------------------
__global__ __launch_bounds__(256)
void lstm_step_kernel(const _Float16* __restrict__ inA1, int ldA1, int K1,
                      const _Float16* __restrict__ hIn,
                      const _Float16* __restrict__ wF, const _Float16* __restrict__ wI,
                      const _Float16* __restrict__ wO, const _Float16* __restrict__ wG,
                      const float* __restrict__ bF, const float* __restrict__ bI,
                      const float* __restrict__ bO, const float* __restrict__ bG,
                      float* __restrict__ cState, _Float16* __restrict__ hOut,
                      int K, int firstStep)
{
    const int lane = threadIdx.x & 31;
    const int wid  = (blockIdx.x * blockDim.x + threadIdx.x) >> 5;  // 0..127
    const int mTile = wid & 3;    // 4 tiles cover B=64 rows
    const int nTile = wid >> 2;   // 32 tiles cover H=512 cols
    const int row0 = mTile * 16;
    const int col0 = nTile * 16;
    const int lh   = lane & 15;
    const int half = lane >> 4;

    const int mrow = row0 + lh;   // A-fragment row (batch index)
    const int ncol = col0 + lh;   // B-fragment col (gate unit index)

    // Accumulators start at bias (bias depends only on column n)
    v8f accF, accI, accO, accG;
    {
        const float vf = bF[ncol], vi = bI[ncol], vo = bO[ncol], vg = bG[ncol];
        #pragma unroll
        for (int v = 0; v < 8; ++v) { accF[v]=vf; accI[v]=vi; accO[v]=vo; accG[v]=vg; }
    }

    const _Float16* __restrict__ aRow1 = inA1 + (size_t)mrow * ldA1;
    const _Float16* __restrict__ aRow2 = hIn  + (size_t)mrow * HH;
    const _Float16* __restrict__ wFr = wF + (size_t)ncol * K;
    const _Float16* __restrict__ wIr = wI + (size_t)ncol * K;
    const _Float16* __restrict__ wOr = wO + (size_t)ncol * K;
    const _Float16* __restrict__ wGr = wG + (size_t)ncol * K;

    // Segment 1: input part of the concat (k in [0, K1))
    for (int kb = 0; kb < K1; kb += 32) {
        Frag a  = load_a_frag(aRow1, kb, half);
        Frag fb = load_b_frag(wFr, kb, half);
        Frag ib = load_b_frag(wIr, kb, half);
        Frag ob = load_b_frag(wOr, kb, half);
        Frag gb = load_b_frag(wGr, kb, half);
        accF = __builtin_amdgcn_wmma_f32_16x16x32_f16(false, a.v, false, fb.v, (short)0, accF, false, false);
        accI = __builtin_amdgcn_wmma_f32_16x16x32_f16(false, a.v, false, ib.v, (short)0, accI, false, false);
        accG = __builtin_amdgcn_wmma_f32_16x16x32_f16(false, a.v, false, gb.v, (short)0, accG, false, false);
        accO = __builtin_amdgcn_wmma_f32_16x16x32_f16(false, a.v, false, ob.v, (short)0, accO, false, false);
    }
    // Segment 2: recurrent part (k in [K1, K)), A from hIn
    for (int kb = K1; kb < K; kb += 32) {
        Frag a  = load_a_frag(aRow2, kb - K1, half);
        Frag fb = load_b_frag(wFr, kb, half);
        Frag ib = load_b_frag(wIr, kb, half);
        Frag ob = load_b_frag(wOr, kb, half);
        Frag gb = load_b_frag(wGr, kb, half);
        accF = __builtin_amdgcn_wmma_f32_16x16x32_f16(false, a.v, false, fb.v, (short)0, accF, false, false);
        accI = __builtin_amdgcn_wmma_f32_16x16x32_f16(false, a.v, false, ib.v, (short)0, accI, false, false);
        accG = __builtin_amdgcn_wmma_f32_16x16x32_f16(false, a.v, false, gb.v, (short)0, accG, false, false);
        accO = __builtin_amdgcn_wmma_f32_16x16x32_f16(false, a.v, false, ob.v, (short)0, accO, false, false);
    }

    // Cell update; D layout: n = lane%16, m = half*8 + v
    #pragma unroll
    for (int v = 0; v < 8; ++v) {
        const int bidx = row0 + (half << 3) + v;
        const size_t idx = (size_t)bidx * HH + ncol;
        float f = fast_sigmoid(accF[v]);
        float i = fast_sigmoid(accI[v]);
        float g = fast_tanh(accG[v]);
        float o = fast_sigmoid(accO[v]);
        float cOld = firstStep ? 0.0f : cState[idx];
        float cNew = f * cOld + i * g;
        cState[idx] = cNew;
        hOut[idx]   = (_Float16)(o * fast_tanh(cNew));
    }
}

// ---------------------------------------------------------------------------
// FC head (tiny: 34 MFLOP total) — plain FMA kernels
// ---------------------------------------------------------------------------
__global__ void fc1_kernel(const _Float16* __restrict__ h,
                           const float* __restrict__ w, const float* __restrict__ b,
                           float* __restrict__ out) {
    int idx = blockIdx.x * blockDim.x + threadIdx.x;   // 64*512 threads
    int bi = idx >> 9, j = idx & 511;
    const _Float16* hr = h + (size_t)bi * HH;
    const float*    wr = w + (size_t)j * HH;
    float s = b[j];
    #pragma unroll 8
    for (int k = 0; k < HH; ++k) s = fmaf((float)hr[k], wr[k], s);
    out[idx] = s;
}

__global__ void fc2_relu_kernel(const float* __restrict__ hid,
                                const float* __restrict__ w, const float* __restrict__ b,
                                float* __restrict__ out) {
    int idx = blockIdx.x * blockDim.x + threadIdx.x;   // 64*24 threads
    if (idx >= BB * OUTN) return;
    int bi = idx / OUTN, o = idx % OUTN;
    const float* hr = hid + (size_t)bi * HH;
    const float* wr = w + (size_t)o * HH;
    float s = b[o];
    #pragma unroll 8
    for (int k = 0; k < HH; ++k) s = fmaf(hr[k], wr[k], s);
    out[idx] = fmaxf(s, 0.0f);
}

// ---------------------------------------------------------------------------
// Host driver
// in order: 0:x 1:wf0 2:bf0 3:wi0 4:bi0 5:wo0 6:bo0 7:wg0 8:bg0
//           9:wf1 10:bf1 11:wi1 12:bi1 13:wo1 14:bo1 15:wg1 16:bg1
//           17:fc1_w 18:fc1_b 19:fc2_w 20:fc2_b
// ---------------------------------------------------------------------------
extern "C" void kernel_launch(void* const* d_in, const int* in_sizes, int n_in,
                              void* d_out, int out_size, void* d_ws, size_t ws_size,
                              hipStream_t stream) {
    (void)in_sizes; (void)n_in; (void)out_size; (void)ws_size;

    const float* x      = (const float*)d_in[0];
    const float* wf32[8] = { (const float*)d_in[1], (const float*)d_in[3],
                             (const float*)d_in[5], (const float*)d_in[7],
                             (const float*)d_in[9], (const float*)d_in[11],
                             (const float*)d_in[13], (const float*)d_in[15] };
    const float* bias[8] = { (const float*)d_in[2], (const float*)d_in[4],
                             (const float*)d_in[6], (const float*)d_in[8],
                             (const float*)d_in[10], (const float*)d_in[12],
                             (const float*)d_in[14], (const float*)d_in[16] };
    const float* fc1w = (const float*)d_in[17];
    const float* fc1b = (const float*)d_in[18];
    const float* fc2w = (const float*)d_in[19];
    const float* fc2b = (const float*)d_in[20];

    // Workspace carve-out (256B aligned slices)
    char* ws = (char*)d_ws;
    size_t off = 0;
    auto carve = [&](size_t bytes) -> char* {
        char* p = ws + off;
        off += (bytes + 255) & ~(size_t)255;
        return p;
    };
    const size_t nX   = (size_t)BB * SS * DD;        // 4,194,304
    const size_t nW0  = (size_t)HH * (DD + HH);      // 327,680 per gate
    const size_t nW1  = (size_t)HH * (2 * HH);       // 524,288 per gate
    const size_t nHC  = (size_t)BB * HH;             // 32,768

    _Float16* x16 = (_Float16*)carve(nX * 2);
    _Float16* w16[8];
    for (int g = 0; g < 4; ++g) w16[g]     = (_Float16*)carve(nW0 * 2);
    for (int g = 0; g < 4; ++g) w16[4 + g] = (_Float16*)carve(nW1 * 2);
    _Float16* h0buf[2] = { (_Float16*)carve(nHC * 2), (_Float16*)carve(nHC * 2) };
    _Float16* h1buf[2] = { (_Float16*)carve(nHC * 2), (_Float16*)carve(nHC * 2) };
    float* c0  = (float*)carve(nHC * 4);
    float* c1  = (float*)carve(nHC * 4);
    float* hid = (float*)carve(nHC * 4);

    // One-time (per call) precision conversion: f32 -> f16
    convert_f32_to_f16<<<2048, 256, 0, stream>>>(x, x16, (int)nX);
    for (int g = 0; g < 4; ++g)
        convert_f32_to_f16<<<640, 256, 0, stream>>>(wf32[g], w16[g], (int)nW0);
    for (int g = 0; g < 4; ++g)
        convert_f32_to_f16<<<1024, 256, 0, stream>>>(wf32[4 + g], w16[4 + g], (int)nW1);
    // h state read at t=0 must be zero (c handled by firstStep flag)
    zero_f16<<<128, 256, 0, stream>>>(h0buf[0], h1buf[0], (int)nHC);

    // Sequential scan: 2 fused step-kernels per timestep; double-buffered h
    for (int t = 0; t < SS; ++t) {
        const _Float16* h0r = h0buf[t & 1];
        _Float16*       h0w = h0buf[(t + 1) & 1];
        const _Float16* h1r = h1buf[t & 1];
        _Float16*       h1w = h1buf[(t + 1) & 1];
        const int first = (t == 0) ? 1 : 0;

        // Layer 0: comb = [x_t (K1=128) | h0], K=640
        lstm_step_kernel<<<16, 256, 0, stream>>>(
            x16 + (size_t)t * DD, SS * DD, DD, h0r,
            w16[0], w16[1], w16[2], w16[3],
            bias[0], bias[1], bias[2], bias[3],
            c0, h0w, DD + HH, first);

        // Layer 1: comb = [h0_new (K1=512) | h1], K=1024
        lstm_step_kernel<<<16, 256, 0, stream>>>(
            h0w, HH, HH, h1r,
            w16[4], w16[5], w16[6], w16[7],
            bias[4], bias[5], bias[6], bias[7],
            c1, h1w, 2 * HH, first);
    }

    // After 512 steps, final h1 lives in buffer (512 & 1) == 0
    const _Float16* hFinal = h1buf[0];
    fc1_kernel<<<(BB * HH) / 256, 256, 0, stream>>>(hFinal, fc1w, fc1b, hid);
    fc2_relu_kernel<<<(BB * OUTN + 255) / 256, 256, 0, stream>>>(hid, fc2w, fc2b, (float*)d_out);
}
